// MultiHeadedAttention_66365834658426
// MI455X (gfx1250) — matrix operand order repaired and checked by reading, compiled, not run on hardware
//
#include <hip/hip_runtime.h>

#define DEV __device__ __forceinline__

typedef __attribute__((ext_vector_type(16))) __bf16 bf16x16;
typedef __attribute__((ext_vector_type(8)))  __bf16 bf16x8;
typedef __attribute__((ext_vector_type(8)))  float  v8f;
typedef __attribute__((ext_vector_type(4)))  unsigned int u32x4;
typedef __attribute__((ext_vector_type(8)))  unsigned int u32x8;

constexpr int BATCH = 4, SEQ = 2048, DIM = 1024, NH = 16, DK = 64;
constexpr int MROWS = BATCH * SEQ;   // 8192

DEV v8f wmma_bf16(bf16x16 a, bf16x16 b, v8f c) {
  return __builtin_amdgcn_wmma_f32_16x16x32_bf16(false, a, false, b, (short)0, c,
                                                 false, false);
}

DEV void cvt4(bf16x16& f, int base, float4 v) {
  f[base + 0] = (__bf16)v.x; f[base + 1] = (__bf16)v.y;
  f[base + 2] = (__bf16)v.z; f[base + 3] = (__bf16)v.w;
}

// A fragment (16x32, M x K) from f32 row-major, rows mbase..mbase+15, K window [kb,kb+32)
DEV bf16x16 load_a_f32(const float* __restrict__ p, int ld, int mbase, int kb, int lane) {
  int m = mbase + (lane & 15);
  int off = (lane >> 4) * 8;
  const float* r = p + (size_t)m * ld + kb + off;
  bf16x16 f;
  cvt4(f, 0,  *(const float4*)(r));
  cvt4(f, 4,  *(const float4*)(r + 4));
  cvt4(f, 8,  *(const float4*)(r + 16));
  cvt4(f, 12, *(const float4*)(r + 20));
  return f;
}

// B fragment (32x16, K x N) from f32 row-major [N,K]
DEV bf16x16 load_b_f32(const float* __restrict__ p, int ld, int nbase, int kb, int lane) {
  int n = nbase + (lane & 15);
  int k = kb + (lane >> 4) * 8;
  const float* r = p + (size_t)n * ld + k;
  bf16x16 f;
  cvt4(f, 0,  *(const float4*)(r));
  cvt4(f, 4,  *(const float4*)(r + 4));
  cvt4(f, 8,  *(const float4*)(r + 16));
  cvt4(f, 12, *(const float4*)(r + 20));
  return f;
}

// A fragment from bf16 row-major
DEV bf16x16 load_a_bf16(const __bf16* __restrict__ p, int ld, int mbase, int kb, int lane) {
  int m = mbase + (lane & 15);
  int off = (lane >> 4) * 8;
  const __bf16* r = p + (size_t)m * ld + kb + off;
  bf16x8 lo = *(const bf16x8*)(r);
  bf16x8 hi = *(const bf16x8*)(r + 16);
  bf16x16 f;
#pragma unroll
  for (int i = 0; i < 8; ++i) { f[i] = lo[i]; f[i + 8] = hi[i]; }
  return f;
}

// B fragment: 8 bf16 at r and 8 bf16 at r+16 (lane offsets already applied)
DEV bf16x16 load_b16(const __bf16* r) {
  bf16x8 lo = *(const bf16x8*)(r);
  bf16x8 hi = *(const bf16x8*)(r + 16);
  bf16x16 f;
#pragma unroll
  for (int i = 0; i < 8; ++i) { f[i] = lo[i]; f[i + 8] = hi[i]; }
  return f;
}

// ---- Tensor Data Mover: 2D bf16 tile (tile_d0 contiguous elems x tile_d1 rows)
// from global (row stride = stride_elems) into LDS at lds_off. D# layout per
// CDNA5 ISA. Issued per-wave (EXEC ignored).
DEV void tdm_load_2d(unsigned lds_off, const void* gaddr,
                     unsigned tile_d0, unsigned tile_d1,
                     unsigned tens_d0, unsigned tens_d1, unsigned stride_elems) {
  unsigned long long ga = (unsigned long long)(uintptr_t)gaddr;
  u32x4 g0;
  g0[0] = 1u;                                   // count=1, user descriptor
  g0[1] = lds_off;                              // LDS byte address
  g0[2] = (unsigned)(ga & 0xffffffffu);         // global_addr[31:0]
  g0[3] = (unsigned)((ga >> 32) & 0x1ffffffu) | (2u << 30);  // addr[56:32] | type=2
  u32x8 g1;
  g1[0] = (1u << 16);                           // workgroup_mask=0, data_size=1 (2B)
  g1[1] = (tens_d0 & 0xffffu) << 16;            // tensor_dim0[15:0] @ bits 63:48
  g1[2] = ((tens_d0 >> 16) & 0xffffu) | ((tens_d1 & 0xffffu) << 16);
  g1[3] = ((tens_d1 >> 16) & 0xffffu) | ((tile_d0 & 0xffffu) << 16);
  g1[4] = (tile_d1 & 0xffffu);                  // tile_dim1; tile_dim2=0
  g1[5] = stride_elems;                         // tensor_dim0_stride[31:0]
  g1[6] = 0u;                                   // stride[47:32], dim1_stride lo
  g1[7] = 0u;
  asm volatile("tensor_load_to_lds %0, %1" :: "s"(g0), "s"(g1) : "memory");
}

DEV unsigned lds_offset_of(const void* p) {
  return (unsigned)(uintptr_t)p;  // flat->LDS: low 32 bits of the address
}

// ---------------- Q/K/V projection: out = X @ W.T (bf16 out, head-split layout)
__global__ __launch_bounds__(256) void proj_gemm(const float* __restrict__ X,
                                                 const float* __restrict__ W,
                                                 __bf16* __restrict__ out, int vtrans) {
  int lane = threadIdx.x & 31, wave = threadIdx.x >> 5;
  int wm = wave & 3, wn = wave >> 2;
  int mbase = blockIdx.y * 128 + wm * 32;
  int nbase = blockIdx.x * 128 + wn * 64;
  v8f acc[2][4] = {};
  for (int kb = 0; kb < DIM; kb += 32) {
    if (kb + 64 < DIM) {  // prefetch next panel into caches
      __builtin_prefetch(X + (size_t)(mbase + (lane & 15)) * DIM + kb + 64, 0, 3);
      __builtin_prefetch(W + (size_t)(nbase + (lane & 15)) * DIM + kb + 64, 0, 3);
    }
    bf16x16 a0 = load_a_f32(X, DIM, mbase, kb, lane);
    bf16x16 a1 = load_a_f32(X, DIM, mbase + 16, kb, lane);
#pragma unroll
    for (int ns = 0; ns < 4; ++ns) {
      bf16x16 bf = load_b_f32(W, DIM, nbase + ns * 16, kb, lane);
      acc[0][ns] = wmma_bf16(a0, bf, acc[0][ns]);
      acc[1][ns] = wmma_bf16(a1, bf, acc[1][ns]);
    }
  }
  int halfrow = 8 * (lane >> 4);
  int col = lane & 15;
#pragma unroll
  for (int ms = 0; ms < 2; ++ms)
#pragma unroll
    for (int ns = 0; ns < 4; ++ns)
#pragma unroll
      for (int r = 0; r < 8; ++r) {
        int m = mbase + ms * 16 + r + halfrow;
        int n = nbase + ns * 16 + col;
        int b = m >> 11, s = m & (SEQ - 1);
        int h = n >> 6, d = n & (DK - 1);
        size_t idx = vtrans ? ((size_t)((b * NH + h) * DK + d) * SEQ + s)
                            : ((size_t)((b * NH + h) * SEQ + s) * DK + d);
        out[idx] = (__bf16)acc[ms][ns][r];
      }
}

// ---------------- flash attention: 8 waves share one (b,h); K/V staged via TDM
__global__ __launch_bounds__(256) void attn_kernel(const __bf16* __restrict__ Q,
                                                   const __bf16* __restrict__ K,
                                                   const __bf16* __restrict__ Vt,
                                                   const unsigned char* __restrict__ mask,
                                                   __bf16* __restrict__ Xout) {
  __shared__ __attribute__((aligned(16))) __bf16 ktile[2][32 * 64];  // [key][d]
  __shared__ __attribute__((aligned(16))) __bf16 vtile[2][64 * 32];  // [d][key]
  __shared__ __attribute__((aligned(16))) __bf16 psh[8][16 * 32];

  int lane = threadIdx.x & 31, wave = threadIdx.x >> 5;
  int bh = blockIdx.x >> 4;            // 0..63
  int qblk = blockIdx.x & 15;          // SEQ/128
  int b = bh >> 4, h = bh & 15;
  int qbase = qblk * 128 + wave * 16;

  const __bf16* Qp = Q + (size_t)bh * SEQ * DK;
  const __bf16* Kp = K + (size_t)bh * SEQ * DK;
  const __bf16* Vp = Vt + (size_t)bh * DK * SEQ;
  const unsigned char* Mp = mask + (size_t)b * SEQ * SEQ;

  bf16x16 aq[2];
  aq[0] = load_a_bf16(Qp, DK, qbase, 0, lane);
  aq[1] = load_a_bf16(Qp, DK, qbase, 32, lane);

  v8f acc[4] = {};
  float mst[8], lst[8];
#pragma unroll
  for (int r = 0; r < 8; ++r) { mst[r] = -3.0e38f; lst[r] = 0.f; }

  int halfrow = 8 * (lane >> 4);
  int col = lane & 15;
  int kk16 = 8 * (lane >> 4);

  constexpr int NIT = SEQ / 32;  // 64
  if (wave == 0) {  // kick off first K/V tile DMA
    tdm_load_2d(lds_offset_of(&ktile[0][0]), Kp, DK, 32, DK, SEQ, DK);
    tdm_load_2d(lds_offset_of(&vtile[0][0]), Vp, 32, DK, SEQ, DK, SEQ);
  }

  for (int it = 0; it < NIT; ++it) {
    int kv = it * 32;
    if (wave == 0) __builtin_amdgcn_s_wait_tensorcnt(0);  // buf[it&1] ready
    __syncthreads();
    if (wave == 0 && it + 1 < NIT) {  // overlap next tile DMA with compute
      int nb = (it + 1) & 1, nkv = kv + 32;
      tdm_load_2d(lds_offset_of(&ktile[nb][0]), Kp + (size_t)nkv * DK, DK, 32, DK, SEQ, DK);
      tdm_load_2d(lds_offset_of(&vtile[nb][0]), Vp + nkv, 32, DK, SEQ, DK, SEQ);
    }
    const __bf16* Kl = &ktile[it & 1][0];
    const __bf16* Vl = &vtile[it & 1][0];

    // scores: 16 q x 32 keys (from LDS), f32 accum
    v8f s0 = {}, s1 = {};
#pragma unroll
    for (int dc = 0; dc < 2; ++dc) {
      s0 = wmma_bf16(aq[dc], load_b16(Kl + (size_t)col * DK + dc * 32 + kk16), s0);
      s1 = wmma_bf16(aq[dc], load_b16(Kl + (size_t)(16 + col) * DK + dc * 32 + kk16), s1);
    }
    // scale + mask + online softmax (row r: q = qbase + r + halfrow)
#pragma unroll
    for (int r = 0; r < 8; ++r) {
      int q = qbase + r + halfrow;
      const unsigned char* mr = Mp + (size_t)q * SEQ + kv;
      float sv0 = mr[col]      ? -1.0e9f : s0[r] * 0.125f;
      float sv1 = mr[16 + col] ? -1.0e9f : s1[r] * 0.125f;
      float v = fmaxf(sv0, sv1);
#pragma unroll
      for (int x = 1; x < 16; x <<= 1) v = fmaxf(v, __shfl_xor(v, x, 32));
      float nm = fmaxf(mst[r], v);
      float corr = __expf(mst[r] - nm);
      mst[r] = nm;
      float p0 = __expf(sv0 - nm), p1 = __expf(sv1 - nm);
      float rs = p0 + p1;
#pragma unroll
      for (int x = 1; x < 16; x <<= 1) rs += __shfl_xor(rs, x, 32);
      lst[r] = lst[r] * corr + rs;
#pragma unroll
      for (int t = 0; t < 4; ++t) acc[t][r] *= corr;
      int row = r + halfrow;
      psh[wave][row * 32 + col]      = (__bf16)p0;
      psh[wave][row * 32 + 16 + col] = (__bf16)p1;
    }
    asm volatile("s_wait_dscnt 0" ::: "memory");  // C-layout -> A-layout via LDS
    bf16x16 pa;
    {
      const __bf16* pr = &psh[wave][(lane & 15) * 32 + 8 * (lane >> 4)];
      bf16x8 lo = *(const bf16x8*)(pr);
      bf16x8 hi = *(const bf16x8*)(pr + 16);
#pragma unroll
      for (int i = 0; i < 8; ++i) { pa[i] = lo[i]; pa[i + 8] = hi[i]; }
    }
    // ctx += P @ V  (V tile in LDS, rows are d, contiguous in key)
#pragma unroll
    for (int dt = 0; dt < 4; ++dt)
      acc[dt] = wmma_bf16(pa, load_b16(Vl + (size_t)(dt * 16 + col) * 32 + kk16), acc[dt]);

    __syncthreads();  // all waves done with buf[it&1] before it is refilled
  }

  // normalize and write context in [B,S,D] bf16 layout for the FC GEMM
#pragma unroll
  for (int dt = 0; dt < 4; ++dt)
#pragma unroll
    for (int r = 0; r < 8; ++r) {
      int q = qbase + r + halfrow;
      float val = acc[dt][r] / lst[r];
      size_t idx = ((size_t)(b * SEQ + q)) * DIM + h * DK + dt * 16 + col;
      Xout[idx] = (__bf16)val;
    }
}

// ---------------- FC projection + residual: Y = Xb @ Wfc.T + resid  (f32 out)
__global__ __launch_bounds__(256) void fc_gemm(const __bf16* __restrict__ Xb,
                                               const float* __restrict__ W,
                                               const float* __restrict__ resid,
                                               float* __restrict__ Y) {
  int lane = threadIdx.x & 31, wave = threadIdx.x >> 5;
  int wm = wave & 3, wn = wave >> 2;
  int mbase = blockIdx.y * 128 + wm * 32;
  int nbase = blockIdx.x * 128 + wn * 64;
  v8f acc[2][4] = {};
  for (int kb = 0; kb < DIM; kb += 32) {
    if (kb + 64 < DIM) {
      __builtin_prefetch(Xb + (size_t)(mbase + (lane & 15)) * DIM + kb + 64, 0, 3);
      __builtin_prefetch(W + (size_t)(nbase + (lane & 15)) * DIM + kb + 64, 0, 3);
    }
    bf16x16 a0 = load_a_bf16(Xb, DIM, mbase, kb, lane);
    bf16x16 a1 = load_a_bf16(Xb, DIM, mbase + 16, kb, lane);
#pragma unroll
    for (int ns = 0; ns < 4; ++ns) {
      bf16x16 bf = load_b_f32(W, DIM, nbase + ns * 16, kb, lane);
      acc[0][ns] = wmma_bf16(a0, bf, acc[0][ns]);
      acc[1][ns] = wmma_bf16(a1, bf, acc[1][ns]);
    }
  }
  int halfrow = 8 * (lane >> 4);
  int col = lane & 15;
#pragma unroll
  for (int ms = 0; ms < 2; ++ms)
#pragma unroll
    for (int ns = 0; ns < 4; ++ns)
#pragma unroll
      for (int r = 0; r < 8; ++r) {
        int m = mbase + ms * 16 + r + halfrow;
        int n = nbase + ns * 16 + col;
        size_t idx = (size_t)m * DIM + n;
        Y[idx] = acc[ms][ns][r] + resid[idx];
      }
}

// ---------------- LayerNorm over rows of Y
__global__ __launch_bounds__(256) void ln_kernel(const float* __restrict__ Y,
                                                 float* __restrict__ O) {
  __shared__ float rs[8], rss[8];
  int row = blockIdx.x, tid = threadIdx.x;
  int lane = tid & 31, wave = tid >> 5;
  const float* yr = Y + (size_t)row * DIM;
  float4 v = *(const float4*)(yr + tid * 4);
  float s = v.x + v.y + v.z + v.w;
  float ss = v.x * v.x + v.y * v.y + v.z * v.z + v.w * v.w;
#pragma unroll
  for (int x = 1; x < 32; x <<= 1) { s += __shfl_xor(s, x, 32); ss += __shfl_xor(ss, x, 32); }
  if (lane == 0) { rs[wave] = s; rss[wave] = ss; }
  __syncthreads();
  float ts = 0.f, tss = 0.f;
#pragma unroll
  for (int w = 0; w < 8; ++w) { ts += rs[w]; tss += rss[w]; }
  float mu = ts * (1.0f / DIM);
  float var = tss * (1.0f / DIM) - mu * mu;
  float inv = rsqrtf(var + 1e-5f);
  float4 o;
  o.x = (v.x - mu) * inv; o.y = (v.y - mu) * inv;
  o.z = (v.z - mu) * inv; o.w = (v.w - mu) * inv;
  *(float4*)(O + (size_t)row * DIM + tid * 4) = o;
}

extern "C" void kernel_launch(void* const* d_in, const int* in_sizes, int n_in,
                              void* d_out, int out_size, void* d_ws, size_t ws_size,
                              hipStream_t stream) {
  const float* query = (const float*)d_in[0];
  const float* key   = (const float*)d_in[1];
  const float* value = (const float*)d_in[2];
  const unsigned char* mask = (const unsigned char*)d_in[3];
  const float* Wq  = (const float*)d_in[4];
  const float* Wk  = (const float*)d_in[5];
  const float* Wv  = (const float*)d_in[6];
  const float* Wfc = (const float*)d_in[7];

  size_t nelem = (size_t)MROWS * DIM;  // 8,388,608
  __bf16* qb = (__bf16*)d_ws;
  __bf16* kb = qb + nelem;
  __bf16* vt = kb + nelem;
  __bf16* xb = vt + nelem;
  float*  yb = (float*)(xb + nelem);

  dim3 gg(DIM / 128, MROWS / 128);
  proj_gemm<<<gg, 256, 0, stream>>>(query, Wq, qb, 0);
  proj_gemm<<<gg, 256, 0, stream>>>(key,   Wk, kb, 0);
  proj_gemm<<<gg, 256, 0, stream>>>(value, Wv, vt, 1);
  attn_kernel<<<BATCH * NH * (SEQ / 128), 256, 0, stream>>>(qb, kb, vt, mask, xb);
  fc_gemm<<<gg, 256, 0, stream>>>(xb, Wfc, query, yb);
  ln_kernel<<<MROWS, 256, 0, stream>>>(yb, (float*)d_out);
}
